// MambaBlock_34883724378768
// MI455X (gfx1250) — matrix operand order repaired and checked by reading, not compiled, not run on hardware
//
#include <hip/hip_runtime.h>

// ---- Problem constants (match reference) ----
#define BB   2
#define LL   2048
#define DMM  768
#define DII  1536           // 2*DM
#define NN   16
#define RR   48
#define KPD  64             // dt GEMM K padded 48 -> 64
#define BL   (BB*LL)        // 4096 rows
#define XK   (RR + 2*NN)    // 80, x_proj output width
#define EPSF 1e-5f

typedef __attribute__((ext_vector_type(16))) _Float16 v16h;
typedef __attribute__((ext_vector_type(8)))  float    v8f;

__device__ __forceinline__ float sigf(float x) { return 1.f / (1.f + __expf(-x)); }

// ---------------------------------------------------------------------------
// 1) residual = x + residual ; xn = RMSNorm(residual) * norm_w  (f16 for WMMA)
//    one block per (b,l) row, 256 threads, DM=768 -> 3 elems/thread
// ---------------------------------------------------------------------------
__global__ void add_rmsnorm_kernel(const float* __restrict__ x, const float* __restrict__ rin,
                                   const float* __restrict__ nw,
                                   float* __restrict__ rout, _Float16* __restrict__ xn)
{
    __shared__ float sred[256];
    const int row = blockIdx.x;
    const int tid = threadIdx.x;
    const size_t base = (size_t)row * DMM;
    float v[3];
    float ssq = 0.f;
#pragma unroll
    for (int i = 0; i < 3; ++i) {
        int c = tid + i * 256;
        float t = x[base + c] + rin[base + c];
        v[i] = t;
        ssq += t * t;
    }
    sred[tid] = ssq;
    __syncthreads();
    for (int s = 128; s > 0; s >>= 1) {
        if (tid < s) sred[tid] += sred[tid + s];
        __syncthreads();
    }
    const float scale = rsqrtf(sred[0] * (1.f / (float)DMM) + EPSF);
#pragma unroll
    for (int i = 0; i < 3; ++i) {
        int c = tid + i * 256;
        rout[base + c] = v[i];
        xn[base + c] = (_Float16)(v[i] * scale * nw[c]);
    }
}

// ---------------------------------------------------------------------------
// 2) Weight convert: f32 [K][N] (trans=1) or [N][K] (trans=0) -> f16 [N][Kpad]
//    zero-pads K -> Kpad so GEMM K-loop is a multiple of 32.
// ---------------------------------------------------------------------------
__global__ void wconv_kernel(const float* __restrict__ W, _Float16* __restrict__ Wt,
                             int K, int Nn, int Kpad, int trans)
{
    int idx = blockIdx.x * 256 + threadIdx.x;
    if (idx >= Nn * Kpad) return;
    int k = idx % Kpad;
    int n = idx / Kpad;
    float v = 0.f;
    if (k < K) v = trans ? W[(size_t)k * Nn + n] : W[(size_t)n * K + k];
    Wt[idx] = (_Float16)v;
}

// ---------------------------------------------------------------------------
// 3) WMMA GEMM:  C[M x Nc](f32) = A[M x K](f16, row-major, lda=K)
//                               * Bt[Nc x K](f16, row-major = B^T, ldb=K)
//    Each wave computes NT adjacent 16x16 output tiles along N, sharing one
//    A fragment per 32-deep K-step -> A traffic / NT, NT WMMAs per A load.
//    Fragment layouts per ISA 7.12.2:
//      A (16x32 f16): lane m = lane&15, half = lane>>4,
//        vgpr v holds K = ((v&4)<<2) + half*8 + (v&3)*2 (+1)   [2x b128/lane]
//      B (32x16 f16): element e -> same K as A (table order), col = lane&15 [2x b128/lane]
//      D: vgpr v -> row = v + 8*half, col = lane&15.
//    epilogue==1: softplus(acc + bias[col])  (dt projection)
// ---------------------------------------------------------------------------
template <int NT>
__global__ void wmma_gemm_kernel(const _Float16* __restrict__ A, const _Float16* __restrict__ Bt,
                                 float* __restrict__ C, int M, int Nc, int K,
                                 const float* __restrict__ bias, int epilogue)
{
    const int gwave   = (blockIdx.x * blockDim.x + threadIdx.x) >> 5;
    const int lane    = threadIdx.x & 31;
    const int tilesN  = Nc >> 4;
    const int groupsN = tilesN / NT;
    const int tilesM  = M >> 4;
    if (gwave >= tilesM * groupsN) return;   // wave-uniform: EXEC stays all-ones
    const int tm = gwave / groupsN;
    const int tg = gwave - tm * groupsN;
    const int half = lane >> 4;
    const int l16  = lane & 15;

    const _Float16* arow = A + (size_t)((tm << 4) + l16) * K;
    const _Float16* brow[NT];
#pragma unroll
    for (int j = 0; j < NT; ++j)
        brow[j] = Bt + (size_t)(((tg * NT + j) << 4) + l16) * K;

    v8f acc[NT] = {};

    for (int k0 = 0; k0 < K; k0 += 32) {
        v16h a;
#pragma unroll
        for (int v = 0; v < 8; ++v) {
            const int ka = k0 + ((v & 4) << 2) + (half << 3) + ((v & 3) << 1);
            a[2 * v]     = arow[ka];
            a[2 * v + 1] = arow[ka + 1];
        }
#pragma unroll
        for (int j = 0; j < NT; ++j) {
            v16h b;
#pragma unroll
            for (int v = 0; v < 8; ++v) {
                const int kb = k0 + ((v & 4) << 2) + (half << 3) + ((v & 3) << 1);
                b[2 * v]     = brow[j][kb];
                b[2 * v + 1] = brow[j][kb + 1];
            }
            acc[j] = __builtin_amdgcn_wmma_f32_16x16x32_f16(
                        /*neg_a=*/false, a, /*neg_b=*/false, b,
                        /*c_mod=*/(short)0, acc[j], /*reuse_a=*/false, /*reuse_b=*/false);
        }
    }

#pragma unroll
    for (int j = 0; j < NT; ++j) {
        const int ccol = ((tg * NT + j) << 4) + l16;
        float* cp = C + (size_t)((tm << 4) + (half << 3)) * Nc + ccol;
        if (epilogue == 1) {
            const float bv = bias[ccol];
#pragma unroll
            for (int v = 0; v < 8; ++v) {
                float xv = acc[j][v] + bv;
                cp[(size_t)v * Nc] = (xv > 20.f) ? xv : __logf(1.f + __expf(xv));
            }
        } else {
#pragma unroll
            for (int v = 0; v < 8; ++v) cp[(size_t)v * Nc] = acc[j][v];
        }
    }
}

// ---------------------------------------------------------------------------
// 4) Depthwise conv3 (pad 1) over L + bias + SiLU, reading xc = xz[:, :DI].
//    Output u stored f16 at [b][l][d] (feeds x_proj WMMA and the scan).
// ---------------------------------------------------------------------------
__global__ void conv_silu_kernel(const float* __restrict__ xz, const float* __restrict__ cw,
                                 const float* __restrict__ cb, _Float16* __restrict__ uh)
{
    int idx = blockIdx.x * 256 + threadIdx.x;
    if (idx >= BB * LL * DII) return;
    const int d = idx % DII;
    const int l = (idx / DII) % LL;
    const int b = idx / (DII * LL);
    const float* base = xz + ((size_t)b * LL) * (2 * DII) + d;
    float acc = cb[d];
    const float w0 = cw[d * 3 + 0], w1 = cw[d * 3 + 1], w2 = cw[d * 3 + 2];
    if (l > 0)      acc += w0 * base[(size_t)(l - 1) * (2 * DII)];
    acc += w1 * base[(size_t)l * (2 * DII)];
    if (l < LL - 1) acc += w2 * base[(size_t)(l + 1) * (2 * DII)];
    const float u = acc * sigf(acc);
    uh[idx] = (_Float16)u;  // idx == ((b*LL)+l)*DII + d
}

// ---------------------------------------------------------------------------
// 5) Pack dts (x_dbl[:, :48]) into f16 [BL][64] zero-padded for the dt GEMM.
// ---------------------------------------------------------------------------
__global__ void dts_pack_kernel(const float* __restrict__ xdbl, _Float16* __restrict__ dts)
{
    int idx = blockIdx.x * 256 + threadIdx.x;
    if (idx >= BL * KPD) return;
    const int r = idx & (KPD - 1);
    const int row = idx >> 6;
    dts[idx] = (_Float16)((r < RR) ? xdbl[(size_t)row * XK + r] : 0.f);
}

// ---------------------------------------------------------------------------
// 6) Selective scan. One thread per (b,d); h[16] in registers; 2048 serial
//    steps. delta buffer (dly) is read at t then overwritten with y at t
//    (same thread, same index -> safe aliasing; saves a 25 MB buffer).
//    B_t / C_t are broadcast reads shared by all d -> L0/L2 friendly.
// ---------------------------------------------------------------------------
__global__ void scan_kernel(const float* __restrict__ xdbl, float* __restrict__ dly,
                            const _Float16* __restrict__ uh,
                            const float* __restrict__ Alog, const float* __restrict__ Dv)
{
    int idx = blockIdx.x * 256 + threadIdx.x;
    if (idx >= BB * DII) return;
    const int d = idx % DII;
    const int b = idx / DII;
    float Am[NN], h[NN];
#pragma unroll
    for (int n = 0; n < NN; ++n) { Am[n] = -__expf(Alog[d * NN + n]); h[n] = 0.f; }
    const float Dd = Dv[d];

    const float* xb = xdbl + (size_t)b * LL * XK;
    float* dl = dly + ((size_t)b * LL) * DII + d;
    const _Float16* ub = uh + ((size_t)b * LL) * DII + d;

    for (int t = 0; t < LL; ++t) {
        const float dt = dl[(size_t)t * DII];
        const float ut = (float)ub[(size_t)t * DII];
        if (t + 1 < LL) {                         // global_prefetch_b8 (gfx1250)
            __builtin_prefetch(dl + (size_t)(t + 1) * DII, 0, 3);
            __builtin_prefetch(ub + (size_t)(t + 1) * DII, 0, 3);
        }
        const float* Bt = xb + (size_t)t * XK + RR;
        const float* Ct = Bt + NN;
        const float du = dt * ut;
        float yv = 0.f;
#pragma unroll
        for (int n = 0; n < NN; ++n) {
            const float dA = __expf(dt * Am[n]);
            h[n] = dA * h[n] + du * Bt[n];
            yv += h[n] * Ct[n];
        }
        dl[(size_t)t * DII] = yv + ut * Dd;       // y overwrites delta slot
    }
}

// ---------------------------------------------------------------------------
// 7) LayerNorm over DI + SiLU(z) gate -> g (f16) for the output GEMM.
//    One block per (b,l) row, DI=1536 -> 6 elems/thread.
// ---------------------------------------------------------------------------
__global__ void ln_gate_kernel(const float* __restrict__ y, const float* __restrict__ xz,
                               const float* __restrict__ lw, const float* __restrict__ lb,
                               _Float16* __restrict__ g)
{
    __shared__ float s1[256], s2[256];
    const int row = blockIdx.x;
    const int tid = threadIdx.x;
    const size_t base = (size_t)row * DII;
    float v[6];
    float s = 0.f, sq = 0.f;
#pragma unroll
    for (int i = 0; i < 6; ++i) {
        float t = y[base + tid + i * 256];
        v[i] = t; s += t; sq += t * t;
    }
    s1[tid] = s; s2[tid] = sq;
    __syncthreads();
    for (int st = 128; st > 0; st >>= 1) {
        if (tid < st) { s1[tid] += s1[tid + st]; s2[tid] += s2[tid + st]; }
        __syncthreads();
    }
    const float mu  = s1[0] * (1.f / (float)DII);
    const float var = s2[0] * (1.f / (float)DII) - mu * mu;
    const float rstd = rsqrtf(var + EPSF);
    const float* zrow = xz + (size_t)row * (2 * DII) + DII;
#pragma unroll
    for (int i = 0; i < 6; ++i) {
        int c = tid + i * 256;
        const float yl = (v[i] - mu) * rstd * lw[c] + lb[c];
        const float zv = zrow[c];
        g[base + c] = (_Float16)(yl * (zv * sigf(zv)));
    }
}

// ---------------------------------------------------------------------------
// Orchestration
// ---------------------------------------------------------------------------
extern "C" void kernel_launch(void* const* d_in, const int* in_sizes, int n_in,
                              void* d_out, int out_size, void* d_ws, size_t ws_size,
                              hipStream_t stream)
{
    const float* x    = (const float*)d_in[0];
    const float* res  = (const float*)d_in[1];
    const float* nw   = (const float*)d_in[2];
    const float* Win  = (const float*)d_in[3];   // (DM, 2*DI)
    const float* cw   = (const float*)d_in[4];   // (DI,1,3)
    const float* cb   = (const float*)d_in[5];
    const float* Wxp  = (const float*)d_in[6];   // (DI, R+2N)
    const float* Wdt  = (const float*)d_in[7];   // (DI, R)
    const float* dtb  = (const float*)d_in[8];
    const float* Alog = (const float*)d_in[9];   // (DI, N)
    const float* Dv   = (const float*)d_in[10];
    const float* lw   = (const float*)d_in[11];
    const float* lb   = (const float*)d_in[12];
    const float* Wout = (const float*)d_in[13];  // (DI, DM)
    (void)in_sizes; (void)n_in; (void)out_size; (void)ws_size;

    float* out_p = (float*)d_out;                 // [BL][DM]
    float* res_p = out_p + (size_t)BL * DMM;      // second tuple element

    char* wsp = (char*)d_ws;
    auto alloc = [&](size_t bytes) -> char* {
        char* p = wsp;
        wsp += (bytes + 255) & ~(size_t)255;
        return p;
    };
    _Float16* xn    = (_Float16*)alloc((size_t)BL * DMM * 2);         // 6.3 MB
    _Float16* WinT  = (_Float16*)alloc((size_t)(2 * DII) * DMM * 2);  // 4.7 MB
    _Float16* WoutT = (_Float16*)alloc((size_t)DMM * DII * 2);        // 2.4 MB
    _Float16* WxpT  = (_Float16*)alloc((size_t)XK * DII * 2);         // 0.25 MB
    _Float16* WdtT  = (_Float16*)alloc((size_t)DII * KPD * 2);        // 0.2 MB
    float*    xzb   = (float*)   alloc((size_t)BL * (2 * DII) * 4);   // 50 MB
    _Float16* uhb   = (_Float16*)alloc((size_t)BL * DII * 2);         // 12.6 MB
    float*    xdbl  = (float*)   alloc((size_t)BL * XK * 4);          // 1.3 MB
    _Float16* dts   = (_Float16*)alloc((size_t)BL * KPD * 2);         // 0.5 MB
    float*    dly   = (float*)   alloc((size_t)BL * DII * 4);         // 25 MB (delta -> y)
    _Float16* gbuf  = (_Float16*)alloc((size_t)BL * DII * 2);         // 12.6 MB

    // 1) add + RMSNorm (also emits residual output)
    add_rmsnorm_kernel<<<BL, 256, 0, stream>>>(x, res, nw, res_p, xn);

    // 2) weight conversion / transposition to f16 [N][Kpad]
    wconv_kernel<<<((2 * DII) * DMM + 255) / 256, 256, 0, stream>>>(Win,  WinT,  DMM, 2 * DII, DMM, 1);
    wconv_kernel<<<(DMM * DII + 255) / 256,       256, 0, stream>>>(Wout, WoutT, DII, DMM,     DII, 1);
    wconv_kernel<<<(XK * DII + 255) / 256,        256, 0, stream>>>(Wxp,  WxpT,  DII, XK,      DII, 1);
    wconv_kernel<<<(DII * KPD + 255) / 256,       256, 0, stream>>>(Wdt,  WdtT,  RR,  DII,     KPD, 0);

    // 3) xz = xn @ in_proj   (4096 x 3072, K=768), 4 N-tiles per wave
    {
        int waves = (BL / 16) * ((2 * DII) / 16 / 4);
        wmma_gemm_kernel<4><<<(waves + 7) / 8, 256, 0, stream>>>(xn, WinT, xzb, BL, 2 * DII, DMM, nullptr, 0);
    }

    // 4) depthwise conv + SiLU -> u (f16)
    conv_silu_kernel<<<(BB * LL * DII + 255) / 256, 256, 0, stream>>>(xzb, cw, cb, uhb);

    // 5) x_dbl = u @ x_proj   (4096 x 80, K=1536), all 5 N-tiles per wave
    {
        int waves = (BL / 16) * (XK / 16 / 5);
        wmma_gemm_kernel<5><<<(waves + 7) / 8, 256, 0, stream>>>(uhb, WxpT, xdbl, BL, XK, DII, nullptr, 0);
    }

    // 6) dts pack + delta = softplus(dts @ dt_proj^T + bias)  (4096 x 1536, K=64)
    dts_pack_kernel<<<(BL * KPD + 255) / 256, 256, 0, stream>>>(xdbl, dts);
    {
        int waves = (BL / 16) * (DII / 16 / 4);
        wmma_gemm_kernel<4><<<(waves + 7) / 8, 256, 0, stream>>>(dts, WdtT, dly, BL, DII, KPD, dtb, 1);
    }

    // 7) selective scan (y overwrites delta buffer)
    scan_kernel<<<(BB * DII + 255) / 256, 256, 0, stream>>>(xdbl, dly, uhb, Alog, Dv);

    // 8) LayerNorm + SiLU(z) gate -> g (f16)
    ln_gate_kernel<<<BL, 256, 0, stream>>>(dly, xzb, lw, lb, gbuf);

    // 9) out = g @ out_proj   (4096 x 768, K=1536), 4 N-tiles per wave
    {
        int waves = (BL / 16) * (DMM / 16 / 4);
        wmma_gemm_kernel<4><<<(waves + 7) / 8, 256, 0, stream>>>(gbuf, WoutT, out_p, BL, DMM, DII, nullptr, 0);
    }
}
